// W8A8Linear_78872779424008
// MI455X (gfx1250) — compile-verified, matrix operand order checked by reading, not run on hardware
//
#include <hip/hip_runtime.h>

typedef int v8i __attribute__((ext_vector_type(8)));

#define QMAXF 127.0f
#define EPSF  1e-7f
#define BM 128   // block tile M
#define BN 256   // block tile N  (8 waves: 2 x 4, each wave 64x64)

// ---------------- Kernel 1: per-row dynamic symmetric quantization ----------------
// One block per row of x. 256 threads * 16 floats = K=4096 per row.
__global__ __launch_bounds__(256)
void quant_rows(const float* __restrict__ x, signed char* __restrict__ xq,
                float* __restrict__ xscale, int K) {
  const int m   = blockIdx.x;
  const int tid = threadIdx.x;
  const float4* row4 = (const float4*)(x + (size_t)m * K);

  float4 v[4];
  float amax = 0.0f;
#pragma unroll
  for (int i = 0; i < 4; ++i) {
    v[i] = row4[tid * 4 + i];
    amax = fmaxf(amax, fmaxf(fmaxf(fabsf(v[i].x), fabsf(v[i].y)),
                             fmaxf(fabsf(v[i].z), fabsf(v[i].w))));
  }
  // wave32 reduction
#pragma unroll
  for (int o = 16; o > 0; o >>= 1) amax = fmaxf(amax, __shfl_xor(amax, o, 32));
  __shared__ float red[8];
  if ((tid & 31) == 0) red[tid >> 5] = amax;
  __syncthreads();
  amax = red[0];
#pragma unroll
  for (int i = 1; i < 8; ++i) amax = fmaxf(amax, red[i]);

  const float scale = fmaxf(amax, EPSF) / QMAXF;
  const float inv   = QMAXF / fmaxf(amax, EPSF);
  if (tid == 0) xscale[m] = scale;

  int4 packed;
  int* pp = (int*)&packed;
#pragma unroll
  for (int i = 0; i < 4; ++i) {
    const float* f = (const float*)&v[i];
    int p = 0;
#pragma unroll
    for (int j = 0; j < 4; ++j) {
      int q = (int)rintf(f[j] * inv);           // round-to-nearest-even like jnp.round
      q = max(-127, min(127, q));
      p |= (q & 255) << (8 * j);
    }
    pp[i] = p;
  }
  ((int4*)(xq + (size_t)m * K))[tid] = packed;
}

// ---------------- Kernel 2: repack int32-stored weights to int8 ----------------
__global__ __launch_bounds__(256)
void packw(const int* __restrict__ w, signed char* __restrict__ wq, int total4) {
  const int idx = blockIdx.x * 256 + threadIdx.x;   // each handles 4 source ints
  if (idx >= total4) return;
  int4 w4 = ((const int4*)w)[idx];
  int p = (w4.x & 255) | ((w4.y & 255) << 8) | ((w4.z & 255) << 16) | ((w4.w & 255) << 24);
  ((int*)wq)[idx] = p;
}

// ---------------- int8 16x64 WMMA fragment load (row-major, stride K) ----------------
// ISA 8-bit A/B layout: lanes 0-15 hold row (lane&15) at K-offsets {0..7,16..23,32..39,48..55},
// lanes 16-31 hold the same rows shifted by +8 bytes -> base offset ((lane>>4)<<3),
// then four 8-byte loads at +0,+16,+32,+48.
__device__ __forceinline__ v8i load_frag(const signed char* p) {
  union { v8i v; unsigned long long q[4]; } u;
  u.q[0] = *(const unsigned long long*)(p);
  u.q[1] = *(const unsigned long long*)(p + 16);
  u.q[2] = *(const unsigned long long*)(p + 32);
  u.q[3] = *(const unsigned long long*)(p + 48);
  return u.v;
}

// ---------------- Kernel 3: int8 GEMM via V_WMMA_I32_16X16X64_IU8 ----------------
// Block = 128x256 C-tile, 8 waves arranged 2(M) x 4(N); each wave owns 64x64:
// 4 A-frags x 4 B-frags -> 16 WMMAs per K-step of 64, on only 8 fragment loads.
__global__ __launch_bounds__(256)
void gemm_i8(const signed char* __restrict__ xq, const signed char* __restrict__ wq,
             const float* __restrict__ xscale, const float* __restrict__ wscale,
             const float* __restrict__ bias, float* __restrict__ out,
             int M, int N, int K) {
  const int lane   = threadIdx.x & 31;
  const int wave   = threadIdx.x >> 5;
  const int m_base = blockIdx.y * BM + (wave & 1) * 64;
  const int n_base = blockIdx.x * BN + (wave >> 1) * 64;

  const int row_lo = lane & 15;
  const int koff   = (lane >> 4) << 3;

  const signed char* aptr[4];
  const signed char* bptr[4];
#pragma unroll
  for (int i = 0; i < 4; ++i) {
    aptr[i] = xq + (size_t)(m_base + i * 16 + row_lo) * K + koff;
    bptr[i] = wq + (size_t)(n_base + i * 16 + row_lo) * K + koff;
  }

  v8i acc[4][4];
#pragma unroll
  for (int i = 0; i < 4; ++i)
#pragma unroll
    for (int j = 0; j < 4; ++j) acc[i][j] = (v8i){};

#pragma unroll 2
  for (int k = 0; k < K; k += 64) {
    v8i a[4], b[4];
#pragma unroll
    for (int i = 0; i < 4; ++i) a[i] = load_frag(aptr[i] + k);
#pragma unroll
    for (int j = 0; j < 4; ++j) b[j] = load_frag(bptr[j] + k);
    // signed A x signed B, int32 accumulate
#pragma unroll
    for (int j = 0; j < 4; ++j)
#pragma unroll
      for (int i = 0; i < 4; ++i)
        acc[i][j] = __builtin_amdgcn_wmma_i32_16x16x64_iu8(
            true, a[i], true, b[j], acc[i][j], false, false);
  }

  // Epilogue. C/D layout: n = lane&15 within frag; m = vgpr_idx + 8*(lane>=16).
  float xs[4][8];
#pragma unroll
  for (int i = 0; i < 4; ++i)
#pragma unroll
    for (int r = 0; r < 8; ++r)
      xs[i][r] = xscale[m_base + i * 16 + koff + r];

#pragma unroll
  for (int j = 0; j < 4; ++j) {
    const int n  = n_base + j * 16 + row_lo;
    const float ws = wscale[n];
    const float bs = bias[n];
#pragma unroll
    for (int i = 0; i < 4; ++i) {
      union { v8i v; int s[8]; } u;
      u.v = acc[i][j];
#pragma unroll
      for (int r = 0; r < 8; ++r) {
        const int m = m_base + i * 16 + koff + r;
        out[(size_t)m * N + n] = (float)u.s[r] * xs[i][r] * ws + bs;
      }
    }
  }
}

// ---------------- launch ----------------
extern "C" void kernel_launch(void* const* d_in, const int* in_sizes, int n_in,
                              void* d_out, int out_size, void* d_ws, size_t ws_size,
                              hipStream_t stream) {
  const float* x      = (const float*)d_in[0];
  const int*   w      = (const int*)  d_in[1];
  const float* wscale = (const float*)d_in[2];
  const float* bias   = (const float*)d_in[3];
  float*       out    = (float*)d_out;

  const int N = in_sizes[2];                       // 4096
  const int K = (int)((long long)in_sizes[1] / N); // 4096
  const int M = (int)((long long)in_sizes[0] / K); // 8192

  // workspace layout: x_q [M*K] int8 | w_q [N*K] int8 | x_scale [M] f32
  signed char* xq = (signed char*)d_ws;
  signed char* wq = xq + (size_t)M * K;
  float* xscale   = (float*)(wq + (size_t)N * K);

  quant_rows<<<M, 256, 0, stream>>>(x, xq, xscale, K);

  const int total4 = (int)(((long long)N * K) / 4);
  packw<<<(total4 + 255) / 256, 256, 0, stream>>>(w, wq, total4);

  dim3 grid(N / BN, M / BM);
  gemm_i8<<<grid, 256, 0, stream>>>(xq, wq, xscale, wscale, bias, out, M, N, K);
}
